// RNNEncoder_17738214933226
// MI455X (gfx1250) — matrix operand order repaired and checked by reading, not compiled, not run on hardware
//
#include <hip/hip_runtime.h>
#include <hip/hip_bf16.h>

typedef __attribute__((ext_vector_type(16))) __bf16 v16bf;
typedef __attribute__((ext_vector_type(8)))  __bf16 v8bf;
typedef __attribute__((ext_vector_type(8)))  float  v8f;

// Problem constants
static constexpr int kB   = 8;
static constexpr int kT   = 24;
static constexpr int kCIN = 512;
static constexpr int kHID = 512;
static constexpr int kG   = 4 * kHID;     // 2048 gate channels per dir
static constexpr int kK0  = kCIN * 9;     // 4608 (layer0 im2col K)
static constexpr int kK1  = 2 * kHID * 9; // 9216 (layer1 im2col K)
static constexpr int kN0  = kB * kT * 4;  // 768  (layer0 x-gemm N)
static constexpr int kN1  = kB * kT;      // 192  (layer1 x-gemm N)

// ---------------------------------------------------------------------------
// Fragment loaders (CDNA5 wave32 WMMA 16x16x32 bf16; A/B use a contiguous k order, not the ISA 7.12.2 table)
// A 16x32: lane L holds row M=L%16; K = 16*half + j (contiguous 16)
//          where half = L/16.
// B 32x16 stored transposed (n-major, K contiguous): lane L holds col N=L%16,
//          K = 16*half + j  -> one contiguous 32B load.
// C: VGPR r -> M = 8*half + r, N = L%16.
// ---------------------------------------------------------------------------
__device__ __forceinline__ v16bf frag_a_from_f32(const float* lo, const float* hi) {
  v16bf a;
#pragma unroll
  for (int j = 0; j < 8; ++j) { a[j] = (__bf16)lo[j]; a[j + 8] = (__bf16)hi[j]; }
  return a;
}
__device__ __forceinline__ v16bf frag_a_from_bf16(const __bf16* base) {
  v8bf lo = *reinterpret_cast<const v8bf*>(base);
  v8bf hi = *reinterpret_cast<const v8bf*>(base + 8);
  v16bf a;
#pragma unroll
  for (int j = 0; j < 8; ++j) { a[j] = lo[j]; a[j + 8] = hi[j]; }
  return a;
}

__device__ __forceinline__ float sigm(float x) { return 1.0f / (1.0f + __expf(-x)); }

// ---------------------------------------------------------------------------
// GEMM (x-conv gates): C[m][n] = sum_k A_f32[m][k]*B_bf16[n][k] + bias[m]
// Wave computes 16 x (16*NT).  Ping-pong double-buffered k-loop (2 k-steps
// per iteration, no register rotation copies); loads run one step ahead of
// the WMMAs.  Requires K % 64 == 0 (holds: 4608, 9216, 512).
// ---------------------------------------------------------------------------
template <int NT>
__global__ __launch_bounds__(128) void gemm_xconv(
    const float* __restrict__ A, const __bf16* __restrict__ B,
    const float* __restrict__ bias, float* __restrict__ C, int N, int K) {
  const int lane = threadIdx.x & 31;
  const int wave = threadIdx.x >> 5;
  const int half = lane >> 4;
  const int r16  = lane & 15;
  const int mbase = (blockIdx.y * 4 + wave) * 16;
  const int nbase = blockIdx.x * (16 * NT);

  const float* Arow = A + (size_t)(mbase + r16) * K;
  const __bf16* Brow[NT];
#pragma unroll
  for (int i = 0; i < NT; ++i)
    Brow[i] = B + (size_t)(nbase + i * 16 + r16) * K + half * 16;

  v8f acc[NT];
#pragma unroll
  for (int i = 0; i < NT; ++i) acc[i] = v8f{};

  v16bf a0 = frag_a_from_f32(Arow + half * 16, Arow + 8 + half * 16);
  v16bf a1;
  v16bf b0v[NT], b1v[NT];
#pragma unroll
  for (int i = 0; i < NT; ++i) b0v[i] = *reinterpret_cast<const v16bf*>(Brow[i]);

  const int S = K >> 5;  // number of 32-wide k-steps (even)
  for (int s = 1; s < S - 1; s += 2) {
    const int ka = s * 32, kb = ka + 32;
    a1 = frag_a_from_f32(Arow + ka + half * 16, Arow + ka + 8 + half * 16);
#pragma unroll
    for (int i = 0; i < NT; ++i)
      b1v[i] = *reinterpret_cast<const v16bf*>(Brow[i] + ka);
#pragma unroll
    for (int i = 0; i < NT; ++i)
      acc[i] = __builtin_amdgcn_wmma_f32_16x16x32_bf16(false, a0, false, b0v[i],
                                                       (short)0, acc[i], false, false);
    a0 = frag_a_from_f32(Arow + kb + half * 16, Arow + kb + 8 + half * 16);
#pragma unroll
    for (int i = 0; i < NT; ++i)
      b0v[i] = *reinterpret_cast<const v16bf*>(Brow[i] + kb);
#pragma unroll
    for (int i = 0; i < NT; ++i)
      acc[i] = __builtin_amdgcn_wmma_f32_16x16x32_bf16(false, a1, false, b1v[i],
                                                       (short)0, acc[i], false, false);
  }
  // tail: step S-2 is resident in (a0,b0v); load + run step S-1
  {
    const int ka = (S - 1) * 32;
    a1 = frag_a_from_f32(Arow + ka + half * 16, Arow + ka + 8 + half * 16);
#pragma unroll
    for (int i = 0; i < NT; ++i)
      b1v[i] = *reinterpret_cast<const v16bf*>(Brow[i] + ka);
#pragma unroll
    for (int i = 0; i < NT; ++i)
      acc[i] = __builtin_amdgcn_wmma_f32_16x16x32_bf16(false, a0, false, b0v[i],
                                                       (short)0, acc[i], false, false);
#pragma unroll
    for (int i = 0; i < NT; ++i)
      acc[i] = __builtin_amdgcn_wmma_f32_16x16x32_bf16(false, a1, false, b1v[i],
                                                       (short)0, acc[i], false, false);
  }

#pragma unroll
  for (int i = 0; i < NT; ++i) {
#pragma unroll
    for (int r = 0; r < 8; ++r) {
      const int m = mbase + half * 8 + r;
      C[(size_t)m * N + nbase + i * 16 + r16] = acc[i][r] + bias[m];
    }
  }
}

// ---------------------------------------------------------------------------
// Recurrent step GEMM: gates[m][n] = sum_k Wh_bf16[m][k]*Bh_bf16[dir][n][k]
//                                     + xg[m][col(b, t or ridx, sp)]
// ---------------------------------------------------------------------------
template <int NT>
__global__ __launch_bounds__(128) void gemm_step(
    const __bf16* __restrict__ A, const __bf16* __restrict__ B,
    const float* __restrict__ xg, float* __restrict__ C,
    const int* __restrict__ lens, int N, int K, int xg_ld, int t,
    int spatial, int m_per_dir, int n_per_dir, int n_valid) {
  const int lane = threadIdx.x & 31;
  const int wave = threadIdx.x >> 5;
  const int half = lane >> 4;
  const int r16  = lane & 15;
  const int mbase = (blockIdx.y * 4 + wave) * 16;
  const int nbase = blockIdx.x * (16 * NT);
  const int dir   = mbase / m_per_dir;

  const __bf16* Arow = A + (size_t)(mbase + r16) * K;
  const __bf16* Brow[NT];
#pragma unroll
  for (int i = 0; i < NT; ++i)
    Brow[i] = B + (size_t)(dir * n_per_dir + nbase + i * 16 + r16) * K + half * 16;

  v8f acc[NT];
#pragma unroll
  for (int i = 0; i < NT; ++i) acc[i] = v8f{};

  v16bf a0 = frag_a_from_bf16(Arow + half * 16);
  v16bf a1;
  v16bf b0v[NT], b1v[NT];
#pragma unroll
  for (int i = 0; i < NT; ++i) b0v[i] = *reinterpret_cast<const v16bf*>(Brow[i]);

  const int S = K >> 5;
  for (int s = 1; s < S - 1; s += 2) {
    const int ka = s * 32, kb = ka + 32;
    a1 = frag_a_from_bf16(Arow + ka + half * 16);
#pragma unroll
    for (int i = 0; i < NT; ++i)
      b1v[i] = *reinterpret_cast<const v16bf*>(Brow[i] + ka);
#pragma unroll
    for (int i = 0; i < NT; ++i)
      acc[i] = __builtin_amdgcn_wmma_f32_16x16x32_bf16(false, a0, false, b0v[i],
                                                       (short)0, acc[i], false, false);
    a0 = frag_a_from_bf16(Arow + kb + half * 16);
#pragma unroll
    for (int i = 0; i < NT; ++i)
      b0v[i] = *reinterpret_cast<const v16bf*>(Brow[i] + kb);
#pragma unroll
    for (int i = 0; i < NT; ++i)
      acc[i] = __builtin_amdgcn_wmma_f32_16x16x32_bf16(false, a1, false, b1v[i],
                                                       (short)0, acc[i], false, false);
  }
  {
    const int ka = (S - 1) * 32;
    a1 = frag_a_from_bf16(Arow + ka + half * 16);
#pragma unroll
    for (int i = 0; i < NT; ++i)
      b1v[i] = *reinterpret_cast<const v16bf*>(Brow[i] + ka);
#pragma unroll
    for (int i = 0; i < NT; ++i)
      acc[i] = __builtin_amdgcn_wmma_f32_16x16x32_bf16(false, a0, false, b0v[i],
                                                       (short)0, acc[i], false, false);
#pragma unroll
    for (int i = 0; i < NT; ++i)
      acc[i] = __builtin_amdgcn_wmma_f32_16x16x32_bf16(false, a1, false, b1v[i],
                                                       (short)0, acc[i], false, false);
  }

#pragma unroll
  for (int i = 0; i < NT; ++i) {
    const int n = nbase + i * 16 + r16;
    if (n < n_valid) {
      const int b  = n / spatial;
      const int sp = n % spatial;
      int tb = t;
      if (dir == 1) { const int L = lens[b]; tb = (t < L) ? (L - 1 - t) : t; }
      const size_t gcol = (size_t)(b * kT + tb) * spatial + sp;
#pragma unroll
      for (int r = 0; r < 8; ++r) {
        const int m = mbase + half * 8 + r;
        C[(size_t)m * N + n] = acc[i][r] + xg[(size_t)m * xg_ld + gcol];
      }
    }
  }
}

// ---------------------------------------------------------------------------
// im2col builders (all emit bf16, n-major / K-contiguous)
// ---------------------------------------------------------------------------
__global__ void im2col_x0(__bf16* __restrict__ X, const float* __restrict__ emb) {
  const int idx = blockIdx.x * blockDim.x + threadIdx.x;
  if (idx >= kN0 * kK0) return;
  const int k = idx % kK0, n = idx / kK0;
  const int c = k / 9, rr = k % 9, ky = rr / 3, kx = rr % 3;
  const int sp = n & 3, bt = n >> 2;
  const int t = bt % kT, b = bt / kT;
  const int oy = sp >> 1, ox = sp & 1;
  const int iy = 2 * oy - 1 + ky, ix = 2 * ox - 1 + kx;
  float v = 0.0f;
  if ((unsigned)iy < 4u && (unsigned)ix < 4u)
    v = emb[(((size_t)(b * kT + t) * kCIN + c) << 4) + iy * 4 + ix];
  X[idx] = (__bf16)v;
}

// layer1 input = concat(hs_f[b][t], hs_b_raw[b][ridx(b,t)]); 2x2 -> 1x1 s2 p1
__global__ void im2col_x1(__bf16* __restrict__ X, const float* __restrict__ hs0,
                          const int* __restrict__ lens) {
  const int idx = blockIdx.x * blockDim.x + threadIdx.x;
  if (idx >= kN1 * kK1) return;
  const int k = idx % kK1, n = idx / kK1;
  const int c1 = k / 9, rr = k % 9, ky = rr / 3, kx = rr % 3;
  const int b = n / kT, t = n % kT;
  const int iy = ky - 1, ix = kx - 1;
  float v = 0.0f;
  if ((unsigned)iy < 2u && (unsigned)ix < 2u) {
    const int sp = iy * 2 + ix;
    if (c1 < kHID) {
      v = hs0[((((size_t)0 * kB + b) * kT + t) * kHID + c1) * 4 + sp];
    } else {
      const int L = lens[b];
      const int tb = (t < L) ? (L - 1 - t) : t;
      v = hs0[((((size_t)1 * kB + b) * kT + tb) * kHID + (c1 - kHID)) * 4 + sp];
    }
  }
  X[idx] = (__bf16)v;
}

// h (2x2, pad1 stride1) -> Bh0[d][n=b*4+sp][k]
__global__ void im2col_h0(__bf16* __restrict__ Bh, const float* __restrict__ h0) {
  const int idx = blockIdx.x * blockDim.x + threadIdx.x;
  if (idx >= 2 * 32 * kK0) return;
  const int k = idx % kK0, rest = idx / kK0;
  const int n = rest & 31, d = rest >> 5;
  const int c = k / 9, rr = k % 9, ky = rr / 3, kx = rr % 3;
  const int b = n >> 2, sp = n & 3;
  const int oy = sp >> 1, ox = sp & 1;
  const int iy = oy - 1 + ky, ix = ox - 1 + kx;
  float v = 0.0f;
  if ((unsigned)iy < 2u && (unsigned)ix < 2u)
    v = h0[((size_t)(d * kB + b) * kHID + c) * 4 + iy * 2 + ix];
  Bh[idx] = (__bf16)v;
}

// layer1 recurrence: only center tap alive (1x1 spatial); pad N=8 -> 16
__global__ void im2col_h1(__bf16* __restrict__ Bh, const float* __restrict__ h1) {
  const int idx = blockIdx.x * blockDim.x + threadIdx.x;
  if (idx >= 16 * kHID) return;
  const int c = idx & (kHID - 1), n = idx >> 9;
  Bh[idx] = (__bf16)((n < kB) ? h1[(size_t)n * kHID + c] : 0.0f);
}

// ---------------------------------------------------------------------------
// Weight prep & state init
// ---------------------------------------------------------------------------
__global__ void cvt_bf16(__bf16* __restrict__ dst, const float* __restrict__ src, int n) {
  const int i = blockIdx.x * blockDim.x + threadIdx.x;
  if (i < n) dst[i] = (__bf16)src[i];
}
__global__ void cvt_wh1_center(__bf16* __restrict__ dst, const float* __restrict__ wh1) {
  const int i = blockIdx.x * blockDim.x + threadIdx.x;
  if (i < kG * kHID) dst[i] = (__bf16)wh1[(size_t)i * 9 + 4];  // tap (1,1), dir 0
}
__global__ void init_state(float* h0, float* c0, float* h1, float* c1) {
  const int i = blockIdx.x * blockDim.x + threadIdx.x;
  if (i < 2 * kB * kHID * 4) { h0[i] = 0.0f; c0[i] = 0.0f; }
  if (i < kB * kHID)         { h1[i] = 0.0f; c1[i] = 0.0f; }
}

// ---------------------------------------------------------------------------
// LSTM pointwise updates
// ---------------------------------------------------------------------------
__global__ void lstm_point0(const float* __restrict__ g, float* __restrict__ h,
                            float* __restrict__ cs, float* __restrict__ hs,
                            const int* __restrict__ lens, int t) {
  const int idx = blockIdx.x * blockDim.x + threadIdx.x;
  if (idx >= 2 * kB * kHID * 4) return;
  const int sp = idx & 3, c = (idx >> 2) & (kHID - 1), b = (idx >> 11) & 7, d = idx >> 14;
  const float* gd = g + (size_t)d * kG * 32;
  const int col = b * 4 + sp;
  const float gi = gd[(size_t)c * 32 + col];
  const float gf = gd[(size_t)(kHID + c) * 32 + col];
  const float gg = gd[(size_t)(2 * kHID + c) * 32 + col];
  const float go = gd[(size_t)(3 * kHID + c) * 32 + col];
  const float cp = cs[idx], hp = h[idx];
  const float cn = sigm(gf) * cp + sigm(gi) * tanhf(gg);
  const float hn = sigm(go) * tanhf(cn);
  const bool m = t < lens[b];
  const float hc = m ? hn : hp, cc = m ? cn : cp;
  h[idx] = hc; cs[idx] = cc;
  hs[((((size_t)d * kB + b) * kT + t) * kHID + c) * 4 + sp] = hc;
}

__global__ void lstm_point1(const float* __restrict__ g, float* __restrict__ h,
                            float* __restrict__ cs, const int* __restrict__ lens, int t) {
  const int idx = blockIdx.x * blockDim.x + threadIdx.x;
  if (idx >= kB * kHID) return;
  const int c = idx & (kHID - 1), b = idx >> 9;
  const float gi = g[(size_t)c * 16 + b];
  const float gf = g[(size_t)(kHID + c) * 16 + b];
  const float gg = g[(size_t)(2 * kHID + c) * 16 + b];
  const float go = g[(size_t)(3 * kHID + c) * 16 + b];
  const float cp = cs[idx], hp = h[idx];
  const float cn = sigm(gf) * cp + sigm(gi) * tanhf(gg);
  const float hn = sigm(go) * tanhf(cn);
  const bool m = t < lens[b];
  h[idx] = m ? hn : hp;
  cs[idx] = m ? cn : cp;
}

// out[b][j] = hT_f[b] . w_out[j][:512] + action[b] . w_out[j][512:515] + b_out[j]
__global__ void final_out(const float* __restrict__ h1, const float* __restrict__ action,
                          const float* __restrict__ w_out, const float* __restrict__ b_out,
                          float* __restrict__ out) {
  const int idx = blockIdx.x * blockDim.x + threadIdx.x;
  if (idx >= kB * 512) return;
  const int b = idx / 512, j = idx % 512;
  const float* wr = w_out + (size_t)j * (kHID + 3);
  float s = b_out[j];
  for (int c = 0; c < kHID; ++c) s += h1[(size_t)b * kHID + c] * wr[c];
  for (int a = 0; a < 3; ++a)    s += action[b * 3 + a] * wr[kHID + a];
  out[idx] = s;
}

// ---------------------------------------------------------------------------
static inline int cdiv(long long a, int b) { return (int)((a + b - 1) / b); }

extern "C" void kernel_launch(void* const* d_in, const int* in_sizes, int n_in,
                              void* d_out, int out_size, void* d_ws, size_t ws_size,
                              hipStream_t stream) {
  const float* emb    = (const float*)d_in[0];
  const float* action = (const float*)d_in[1];
  const int*   lens   = (const int*)  d_in[2];
  const float* wx0    = (const float*)d_in[3];
  const float* wh0    = (const float*)d_in[4];
  const float* b0     = (const float*)d_in[5];
  const float* wx1    = (const float*)d_in[6];
  const float* wh1    = (const float*)d_in[7];
  const float* b1     = (const float*)d_in[8];
  const float* w_out  = (const float*)d_in[9];
  const float* b_out  = (const float*)d_in[10];
  float* out = (float*)d_out;

  char* w = (char*)d_ws;
  auto alloc = [&](size_t bytes) {
    char* p = w; w += (bytes + 255) & ~(size_t)255; return p;
  };
  __bf16* Wh0bf = (__bf16*)alloc((size_t)2 * kG * kK0 * 2);   // 37.7 MB
  __bf16* Wh1c  = (__bf16*)alloc((size_t)kG * kHID * 2);      //  2.1 MB
  __bf16* X0col = (__bf16*)alloc((size_t)kN0 * kK0 * 2);      //  7.1 MB
  __bf16* X1col = (__bf16*)alloc((size_t)kN1 * kK1 * 2);      //  3.5 MB
  float*  Xg0   = (float*) alloc((size_t)2 * kG * kN0 * 4);   // 12.6 MB
  float*  Xg1   = (float*) alloc((size_t)kG * kN1 * 4);       //  1.6 MB
  __bf16* Bh0   = (__bf16*)alloc((size_t)2 * 32 * kK0 * 2);
  __bf16* Bh1   = (__bf16*)alloc((size_t)16 * kHID * 2);
  float*  g0    = (float*) alloc((size_t)2 * kG * 32 * 4);
  float*  g1    = (float*) alloc((size_t)kG * 16 * 4);
  float*  h0    = (float*) alloc((size_t)2 * kB * kHID * 4 * 4);
  float*  c0    = (float*) alloc((size_t)2 * kB * kHID * 4 * 4);
  float*  h1    = (float*) alloc((size_t)kB * kHID * 4);
  float*  c1    = (float*) alloc((size_t)kB * kHID * 4);
  float*  hs0   = (float*) alloc((size_t)2 * kB * kT * kHID * 4 * 4); // 3.1 MB

  const int nWh0 = 2 * kG * kK0;
  cvt_bf16<<<cdiv(nWh0, 256), 256, 0, stream>>>(Wh0bf, wh0, nWh0);
  cvt_wh1_center<<<cdiv(kG * kHID, 256), 256, 0, stream>>>(Wh1c, wh1);
  init_state<<<cdiv(2 * kB * kHID * 4, 256), 256, 0, stream>>>(h0, c0, h1, c1);

  // Layer 0 input gates (both dirs): Xg0[4096 x 768]
  im2col_x0<<<cdiv((long long)kN0 * kK0, 256), 256, 0, stream>>>(X0col, emb);
  gemm_xconv<4><<<dim3(kN0 / 64, (2 * kG) / 64), 128, 0, stream>>>(
      wx0, X0col, b0, Xg0, kN0, kK0);

  // Layer 0 bidirectional recurrence
  for (int t = 0; t < kT; ++t) {
    im2col_h0<<<cdiv(2 * 32 * kK0, 256), 256, 0, stream>>>(Bh0, h0);
    gemm_step<2><<<dim3(1, (2 * kG) / 64), 128, 0, stream>>>(
        Wh0bf, Bh0, Xg0, g0, lens, 32, kK0, kN0, t, 4, kG, 32, 32);
    lstm_point0<<<cdiv(2 * kB * kHID * 4, 256), 256, 0, stream>>>(
        g0, h0, c0, hs0, lens, t);
  }

  // Layer 1 (forward dir only — backward output is unused)
  im2col_x1<<<cdiv((long long)kN1 * kK1, 256), 256, 0, stream>>>(X1col, hs0, lens);
  gemm_xconv<4><<<dim3(kN1 / 64, kG / 64), 128, 0, stream>>>(
      wx1, X1col, b1, Xg1, kN1, kK1);

  for (int t = 0; t < kT; ++t) {
    im2col_h1<<<cdiv(16 * kHID, 256), 256, 0, stream>>>(Bh1, h1);
    gemm_step<1><<<dim3(1, kG / 64), 128, 0, stream>>>(
        Wh1c, Bh1, Xg1, g1, lens, 16, kHID, kN1, t, 1, kG, 16, kB);
    lstm_point1<<<cdiv(kB * kHID, 256), 256, 0, stream>>>(g1, h1, c1, lens, t);
  }

  final_out<<<cdiv(kB * 512, 256), 256, 0, stream>>>(h1, action, w_out, b_out, out);
}
